// GQA_37082747633820
// MI455X (gfx1250) — matrix operand order repaired and checked by reading, compiled, not run on hardware
//
#include <hip/hip_runtime.h>
#include <math.h>

// ---------------- problem constants ----------------
#define Bz   4
#define Nz   1024
#define Dz   2048
#define HQ   16
#define HKV  4
#define DH   128
#define EQKV 3072   // DH*(HQ+2*HKV)

// ---------------- WMMA types ----------------
typedef __bf16 v16bf __attribute__((ext_vector_type(16)));
typedef float  v8f   __attribute__((ext_vector_type(8)));

union Frag { unsigned u[8]; v16bf v; };

__device__ __forceinline__ unsigned short f2bf(float f) {
  unsigned u = __float_as_uint(f);
  unsigned r = u + 0x7fffu + ((u >> 16) & 1u);   // RNE
  return (unsigned short)(r >> 16);
}
__device__ __forceinline__ float bf2f(unsigned short h) {
  return __uint_as_float(((unsigned)h) << 16);
}
__device__ __forceinline__ unsigned pack2(unsigned short e0, unsigned short e1) {
  return (unsigned)e0 | ((unsigned)e1 << 16);
}
// A-matrix 16x32 bf16 layout: VGPR v, lane half -> k offset (plus 8*half)
__device__ __forceinline__ int ak_off(int v) { return (v < 4) ? 2 * v : 2 * v + 8; }

// ---------------- Tensor Data Mover (gfx1250) ----------------
#if defined(__HIP_DEVICE_COMPILE__) && __has_builtin(__builtin_amdgcn_tensor_load_to_lds) && \
    __has_builtin(__builtin_amdgcn_s_wait_tensorcnt)
#define HAVE_TDM 1
#else
#define HAVE_TDM 0
#endif

#if HAVE_TDM
typedef unsigned int u32x4 __attribute__((ext_vector_type(4)));
typedef int          i32x4 __attribute__((ext_vector_type(4)));
typedef int          i32x8 __attribute__((ext_vector_type(8)));

__device__ __forceinline__ unsigned lds_offset(const void* p) {
  return (unsigned)(unsigned long long)(__attribute__((address_space(3))) const void*)p;
}

// 2-D fp32 tile DMA: global (row stride = strideElems) -> LDS (packed tileW x tileH).
// D# per CDNA5 ISA: group0 = {flags, lds_addr, global_addr, type=2},
// group1 = {data_size=4B, tensor_dim0/1, tile_dim0/1, dim0_stride}; groups 2/3 zero (2-D).
__device__ __forceinline__ void tdm_load_2d_f32(const float* gsrc, unsigned ldsOff,
                                                unsigned tileW, unsigned tileH,
                                                unsigned strideElems) {
  unsigned long long ga = (unsigned long long)gsrc;
  u32x4 g0;
  g0[0] = 1u;                                                   // count=1, user mode
  g0[1] = ldsOff;                                               // lds_addr (bytes)
  g0[2] = (unsigned)(ga & 0xffffffffu);                         // global_addr[31:0]
  g0[3] = (unsigned)((ga >> 32) & 0x01ffffffu) | (2u << 30);    // addr[56:32] | type=2
  i32x8 g1;
  g1[0] = (int)(2u << 16);                                      // data_size=2 (4 bytes)
  g1[1] = (int)((tileW & 0xffffu) << 16);                       // tensor_dim0[15:0]
  g1[2] = (int)((tileW >> 16) | ((tileH & 0xffffu) << 16));     // dim0 hi | dim1 lo
  g1[3] = (int)((tileH >> 16) | (tileW << 16));                 // dim1 hi | tile_dim0
  g1[4] = (int)(tileH & 0xffffu);                               // tile_dim1 (tile_dim2=0)
  g1[5] = (int)strideElems;                                     // dim0_stride[31:0]
  g1[6] = 0;                                                    // dim0_stride hi | dim1_stride lo
  g1[7] = 0;
  i32x4 gz = {0, 0, 0, 0};
#if __clang_major__ >= 23
  i32x8 gz8 = {0, 0, 0, 0, 0, 0, 0, 0};
  __builtin_amdgcn_tensor_load_to_lds(g0, g1, gz, gz, gz8, 0);
#else
  __builtin_amdgcn_tensor_load_to_lds(g0, g1, gz, gz, 0);
#endif
}
#endif  // HAVE_TDM

// =====================================================================
// Kernel 1/4: C[M x E] = X[M x K] * W[E x K]^T  with split-bf16 (hi+lo)
// BM=64, BN=64, BK=32; 256 threads = 8 waves; each wave: 1 row strip x 2 col tiles.
// Tiles staged global->LDS via the Tensor Data Mover (TENSORcnt).
// =====================================================================
template <int BM, int BN, int BK>
__global__ __launch_bounds__(256) void gemm_xwT_bf16x2(
    const float* __restrict__ X, const float* __restrict__ W,
    float* __restrict__ C, int M, int K, int E) {
  constexpr int LDT = BK + 2;  // u16 stride; even -> 4B-aligned pair reads
  __shared__ unsigned short aHi[BM * LDT];
  __shared__ unsigned short aLo[BM * LDT];
  __shared__ unsigned short bHi[BN * LDT];
  __shared__ unsigned short bLo[BN * LDT];
#if HAVE_TDM
  __shared__ float sX[BM * BK];  // fp32 staging, written by TDM
  __shared__ float sW[BN * BK];
#endif

  const int tid  = threadIdx.x;
  const int wave = tid >> 5;
  const int lane = tid & 31;
  const int half = lane >> 4;
  const int l16  = lane & 15;
  const int m0   = blockIdx.y * BM;
  const int n0   = blockIdx.x * BN;
  const int rowT = wave & 3;         // 16-row strip within block
  const int colT = (wave >> 2) * 2;  // first of two 16-col tiles

  v8f acc0 = {0, 0, 0, 0, 0, 0, 0, 0};
  v8f acc1 = {0, 0, 0, 0, 0, 0, 0, 0};

  for (int k0 = 0; k0 < K; k0 += BK) {
#if HAVE_TDM
    if (tid == 0) {  // TDM ignores EXEC; one instruction per tile, wave 0 issues
      tdm_load_2d_f32(X + (size_t)m0 * K + k0, lds_offset(sX), BK, BM, (unsigned)K);
      tdm_load_2d_f32(W + (size_t)n0 * K + k0, lds_offset(sW), BK, BN, (unsigned)K);
      __builtin_amdgcn_s_wait_tensorcnt(0);
    }
    __syncthreads();
#endif
    // fp32 -> bf16 hi/lo split
    for (int i = tid; i < BM * BK; i += 256) {
      int m = i / BK, k = i % BK;
#if HAVE_TDM
      float x = sX[i];
#else
      float x = X[(size_t)(m0 + m) * K + k0 + k];
#endif
      unsigned short h = f2bf(x);
      aHi[m * LDT + k] = h;
      aLo[m * LDT + k] = f2bf(x - bf2f(h));
    }
    for (int i = tid; i < BN * BK; i += 256) {
      int n = i / BK, k = i % BK;
#if HAVE_TDM
      float w = sW[i];
#else
      float w = W[(size_t)(n0 + n) * K + k0 + k];
#endif
      unsigned short h = f2bf(w);
      bHi[n * LDT + k] = h;
      bLo[n * LDT + k] = f2bf(w - bf2f(h));
    }
    __syncthreads();

    Frag fah, fal, fb0h, fb0l, fb1h, fb1l;
    const int am = rowT * 16 + l16;
    const int nA = (colT + 0) * 16 + l16;
    const int nB = (colT + 1) * 16 + l16;
#pragma unroll
    for (int v = 0; v < 8; ++v) {
      int ka = ak_off(v) + 8 * half;
      fah.u[v] = *(const unsigned*)&aHi[am * LDT + ka];
      fal.u[v] = *(const unsigned*)&aLo[am * LDT + ka];
      int kb = ak_off(v) + 8 * half;
      fb0h.u[v] = *(const unsigned*)&bHi[nA * LDT + kb];
      fb0l.u[v] = *(const unsigned*)&bLo[nA * LDT + kb];
      fb1h.u[v] = *(const unsigned*)&bHi[nB * LDT + kb];
      fb1l.u[v] = *(const unsigned*)&bLo[nB * LDT + kb];
    }
    // fp32 ~= hi*hi + hi*lo + lo*hi
    acc0 = __builtin_amdgcn_wmma_f32_16x16x32_bf16(false, fah.v, false, fb0h.v, (short)0, acc0, false, false);
    acc0 = __builtin_amdgcn_wmma_f32_16x16x32_bf16(false, fah.v, false, fb0l.v, (short)0, acc0, false, false);
    acc0 = __builtin_amdgcn_wmma_f32_16x16x32_bf16(false, fal.v, false, fb0h.v, (short)0, acc0, false, false);
    acc1 = __builtin_amdgcn_wmma_f32_16x16x32_bf16(false, fah.v, false, fb1h.v, (short)0, acc1, false, false);
    acc1 = __builtin_amdgcn_wmma_f32_16x16x32_bf16(false, fah.v, false, fb1l.v, (short)0, acc1, false, false);
    acc1 = __builtin_amdgcn_wmma_f32_16x16x32_bf16(false, fal.v, false, fb1h.v, (short)0, acc1, false, false);
    __syncthreads();  // protects aHi/... and sX/sW for next iteration
  }

#pragma unroll
  for (int r = 0; r < 8; ++r) {
    int m  = m0 + rowT * 16 + r + 8 * half;
    int nA = n0 + (colT + 0) * 16 + l16;
    int nB = n0 + (colT + 1) * 16 + l16;
    C[(size_t)m * E + nA] = acc0[r];
    C[(size_t)m * E + nB] = acc1[r];
  }
}

// =====================================================================
// Kernel 2/4: RMS-norm + RoPE, fp32 qkv -> bf16 Q[b,hq,n,d], K/V[b,hkv,n,d]
// =====================================================================
__global__ __launch_bounds__(256) void normrope_kernel(
    const float* __restrict__ qkv, const int* __restrict__ pos,
    const float* __restrict__ qw, const float* __restrict__ kw,
    unsigned short* __restrict__ Qb, unsigned short* __restrict__ Kb,
    unsigned short* __restrict__ Vb) {
  const int tid = threadIdx.x;
  const int wave = tid >> 5, lane = tid & 31;
  int g = blockIdx.x * 8 + wave;
  const int h = g % 24; g /= 24;
  const int n = g % Nz;
  const int b = g / Nz;

  const float* row = qkv + ((size_t)(b * Nz + n)) * EQKV + h * DH;
  float x0 = row[lane], x1 = row[lane + 32], x2 = row[lane + 64], x3 = row[lane + 96];

  if (h < HQ + HKV) {  // q or k head: rms-norm + rope
    float ss = x0 * x0 + x1 * x1 + x2 * x2 + x3 * x3;
#pragma unroll
    for (int off = 16; off > 0; off >>= 1) ss += __shfl_xor(ss, off, 32);
    const float eps = 1.1920929e-07f;  // finfo(float32).eps
    float r = rsqrtf(ss * (1.f / 128.f) + eps);
    const float* w = (h < HQ) ? qw : kw;
    float t0 = x0 * r * w[lane];
    float t1 = x1 * r * w[lane + 32];
    float t2 = x2 * r * w[lane + 64];
    float t3 = x3 * r * w[lane + 96];

    float p = (float)pos[n];
    const float lnb_over = 9.210340371976184f / 64.f;  // ln(10000)/64
    float a0 = p * __expf(-lnb_over * (float)lane);
    float a1 = p * __expf(-lnb_over * (float)(lane + 32));
    float s0, c0, s1, c1;
    __sincosf(a0, &s0, &c0);
    __sincosf(a1, &s1, &c1);
    float o0 = t0 * c0 - t2 * s0;  // d = lane
    float o2 = t2 * c0 + t0 * s0;  // d = lane+64
    float o1 = t1 * c1 - t3 * s1;  // d = lane+32
    float o3 = t3 * c1 + t1 * s1;  // d = lane+96

    unsigned short* dst;
    if (h < HQ)
      dst = Qb + ((size_t)(b * HQ + h) * Nz + n) * DH;
    else
      dst = Kb + ((size_t)(b * HKV + (h - HQ)) * Nz + n) * DH;
    dst[lane]      = f2bf(o0);
    dst[lane + 32] = f2bf(o1);
    dst[lane + 64] = f2bf(o2);
    dst[lane + 96] = f2bf(o3);
  } else {  // v head: straight bf16 convert
    unsigned short* dst = Vb + ((size_t)(b * HKV + (h - HQ - HKV)) * Nz + n) * DH;
    dst[lane]      = f2bf(x0);
    dst[lane + 32] = f2bf(x1);
    dst[lane + 64] = f2bf(x2);
    dst[lane + 96] = f2bf(x3);
  }
}

// =====================================================================
// Kernel 3/4: causal GQA attention, one wave per (b, h, 16-row q tile).
// =====================================================================
__global__ __launch_bounds__(32) void attn_fwd(
    const unsigned short* __restrict__ Qb, const unsigned short* __restrict__ Kb,
    const unsigned short* __restrict__ Vb, float* __restrict__ Y) {
  __shared__ float S[16 * Nz];  // 64 KB

  const int lane = threadIdx.x;
  const int half = lane >> 4, l16 = lane & 15;
  int idx = blockIdx.x;
  const int qt = idx & 63; idx >>= 6;
  const int h  = idx & 15; idx >>= 4;
  const int b  = idx;
  const int q0 = qt * 16;
  const int kvCount = q0 + 16;
  const int hkv = h >> 2;  // GROUPS = 4

  const unsigned short* Q  = Qb + ((size_t)(b * HQ + h) * Nz + q0) * DH;
  const unsigned short* Kp = Kb + (size_t)(b * HKV + hkv) * Nz * DH;
  const unsigned short* Vp = Vb + (size_t)(b * HKV + hkv) * Nz * DH;
  const float scale = 0.08838834764831845f;  // 128^-0.5
  const v8f vzero = {0, 0, 0, 0, 0, 0, 0, 0};

  Frag qa[4];
#pragma unroll
  for (int ks = 0; ks < 4; ++ks)
#pragma unroll
    for (int v = 0; v < 8; ++v) {
      int d = ks * 32 + ak_off(v) + 8 * half;
      qa[ks].u[v] = *(const unsigned*)(Q + (size_t)l16 * DH + d);
    }

  for (int j = 0; j * 16 < kvCount; ++j) {
    v8f s8 = vzero;
#pragma unroll
    for (int ks = 0; ks < 4; ++ks) {
      Frag kf;
      int kv = j * 16 + l16;
#pragma unroll
      for (int v = 0; v < 8; ++v) {
        int d = ks * 32 + ak_off(v) + 8 * half;
        kf.u[v] = *(const unsigned*)(Kp + (size_t)kv * DH + d);
      }
      s8 = __builtin_amdgcn_wmma_f32_16x16x32_bf16(false, qa[ks].v, false, kf.v, (short)0, s8, false, false);
    }
#pragma unroll
    for (int r = 0; r < 8; ++r) {
      int m  = r + 8 * half;
      int kv = j * 16 + l16;
      float val = s8[r] * scale;
      if (kv > q0 + m) val = -1e30f;  // causal mask -> exp() == 0
      S[m * Nz + kv] = val;
    }
  }
  __syncthreads();

  for (int m = 0; m < 16; ++m) {
    float mx = -1e30f;
    for (int kv = lane; kv < kvCount; kv += 32) mx = fmaxf(mx, S[m * Nz + kv]);
#pragma unroll
    for (int off = 16; off > 0; off >>= 1) mx = fmaxf(mx, __shfl_xor(mx, off, 32));
    float sum = 0.f;
    for (int kv = lane; kv < kvCount; kv += 32) {
      float e = __expf(S[m * Nz + kv] - mx);
      S[m * Nz + kv] = e;
      sum += e;
    }
#pragma unroll
    for (int off = 16; off > 0; off >>= 1) sum += __shfl_xor(sum, off, 32);
    float inv = 1.f / sum;
    for (int kv = lane; kv < kvCount; kv += 32) S[m * Nz + kv] *= inv;
  }
  __syncthreads();

  v8f o[8];
#pragma unroll
  for (int g = 0; g < 8; ++g) o[g] = vzero;

  for (int kv0 = 0; kv0 < kvCount; kv0 += 32) {
    Frag pa;
#pragma unroll
    for (int v = 0; v < 8; ++v) {
      int kv = kv0 + ak_off(v) + 8 * half;
      float p0 = (kv     < kvCount) ? S[l16 * Nz + kv]     : 0.f;
      float p1 = (kv + 1 < kvCount) ? S[l16 * Nz + kv + 1] : 0.f;
      pa.u[v] = pack2(f2bf(p0), f2bf(p1));
    }
#pragma unroll
    for (int g = 0; g < 8; ++g) {
      Frag vf;
      int d = g * 16 + l16;
#pragma unroll
      for (int v = 0; v < 8; ++v) {
        int kv = kv0 + ak_off(v) + 8 * half;
        unsigned short e0 = (kv     < kvCount) ? Vp[(size_t)kv * DH + d]       : (unsigned short)0;
        unsigned short e1 = (kv + 1 < kvCount) ? Vp[(size_t)(kv + 1) * DH + d] : (unsigned short)0;
        vf.u[v] = pack2(e0, e1);
      }
      o[g] = __builtin_amdgcn_wmma_f32_16x16x32_bf16(false, pa.v, false, vf.v, (short)0, o[g], false, false);
    }
  }

#pragma unroll
  for (int g = 0; g < 8; ++g)
#pragma unroll
    for (int r = 0; r < 8; ++r) {
      int m = r + 8 * half;
      int d = g * 16 + l16;
      Y[((size_t)(b * Nz + q0 + m)) * Dz + h * DH + d] = o[g][r];
    }
}

// =====================================================================
// Launch pipeline
// =====================================================================
extern "C" void kernel_launch(void* const* d_in, const int* in_sizes, int n_in,
                              void* d_out, int out_size, void* d_ws, size_t ws_size,
                              hipStream_t stream) {
  const float* x     = (const float*)d_in[0];
  // d_in[1] = mask (bool) -- causal structure implemented directly
  const int*   pos   = (const int*)d_in[2];
  const float* w_qkv = (const float*)d_in[3];
  const float* w_out = (const float*)d_in[4];
  const float* qw    = (const float*)d_in[5];
  const float* kw    = (const float*)d_in[6];
  float* out = (float*)d_out;

  char* ws = (char*)d_ws;
  size_t off = 0;
  float* qkv = (float*)(ws + off);                  off += (size_t)Bz * Nz * EQKV * 4;
  unsigned short* Qb = (unsigned short*)(ws + off); off += (size_t)Bz * HQ  * Nz * DH * 2;
  unsigned short* Kb = (unsigned short*)(ws + off); off += (size_t)Bz * HKV * Nz * DH * 2;
  unsigned short* Vb = (unsigned short*)(ws + off); off += (size_t)Bz * HKV * Nz * DH * 2;
  float* Y = (float*)(ws + off);                    off += (size_t)Bz * Nz * Dz * 4;

  const int M = Bz * Nz;  // 4096

  // 1) qkv = x @ w_qkv^T   (4096 x 3072, K=2048)
  gemm_xwT_bf16x2<64, 64, 32><<<dim3(EQKV / 64, M / 64), 256, 0, stream>>>(
      x, w_qkv, qkv, M, Dz, EQKV);

  // 2) rms-norm + rope -> bf16 Q/K/V
  normrope_kernel<<<(Bz * Nz * 24) / 8, 256, 0, stream>>>(qkv, pos, qw, kw, Qb, Kb, Vb);

  // 3) causal GQA attention -> Y [b,n,h,d] fp32
  attn_fwd<<<Bz * HQ * (Nz / 16), 32, 0, stream>>>(Qb, Kb, Vb, Y);

  // 4) out = Y @ w_out^T   (4096 x 2048, K=2048)
  gemm_xwT_bf16x2<64, 64, 32><<<dim3(Dz / 64, M / 64), 256, 0, stream>>>(
      Y, w_out, out, M, Dz, Dz);
}
